// Encoder_14328010899798
// MI455X (gfx1250) — compile-verified, matrix operand order checked by reading, not run on hardware
//
#include <hip/hip_runtime.h>
#include <stdint.h>

// ---------------------------------------------------------------------------
// MI455X (gfx1250, wave32) implementation of the segmented softmax encoder.
//
//   key0  = x @ W_map + b           (WMMA f16 GEMM, f32 out)
//   keys  = seg_softmax(key0)       (per-segment, batch sorted -> ranges)
//   y     = mlp2(x, v..) * mlp2(keys, k..)    (GEMM+LN+Mish fused, product
//   z     = seg_sum(y)                         fused into 2nd GEMM epilogue)
//   ... same again on keys for keyenc.
//
// All GEMM A-inputs are kept in f16 (WMMA operand format, half the HBM
// traffic); accumulation and softmax/LN statistics stay f32.
// ---------------------------------------------------------------------------

typedef _Float16 v16h __attribute__((ext_vector_type(16)));
typedef float    v8f  __attribute__((ext_vector_type(8)));

#define DIMF   128
#define WLD    136      // padded LDS row stride in halves (272B = 68 banks -> conflict free)
#define NBATCH 4096

union F16Frag { v16h v; uint4 q[2]; };

// A fragment (16x32 f16, row-major global): lane r=lane&15 holds row M=r,
// K = hi*8 + 0..7 (VGPR0-3) and K = 16 + hi*8 + 0..7 (VGPR4-7), hi=lane>>4.
__device__ inline v16h load_a_g(const _Float16* __restrict__ A, int row0, int k0,
                                int nrows, int lane) {
  int r = lane & 15, hi = lane >> 4;
  int row = row0 + r; if (row >= nrows) row = nrows - 1;   // clamp tail reads
  const _Float16* p = A + (size_t)row * DIMF + k0 + hi * 8;
  F16Frag u;
  u.q[0] = *(const uint4*)(p);        // K = k0+hi*8 .. +7     (16B)
  u.q[1] = *(const uint4*)(p + 16);   // K = k0+16+hi*8 .. +7  (16B)
  return u.v;
}

// B fragment (32x16 f16) from LDS-resident W^T (column-major weights):
// same packing as A but indexed by output column.
__device__ inline v16h load_b_lds(const _Float16* Wsh, int n0, int k0, int lane) {
  int c = lane & 15, hi = lane >> 4;
  const _Float16* p = Wsh + (n0 + c) * WLD + k0 + hi * 8;
  F16Frag u;
  u.q[0] = *(const uint4*)(p);
  u.q[1] = *(const uint4*)(p + 16);
  return u.v;
}

// ---------------------------------------------------------------------------
// Fused GEMM: out[rows,128] = A[rows,128](f16) @ W[128,128] + bias, epilogue:
//   EPI=0: write f32
//   EPI=1: LayerNorm(gamma,beta) + Mish, write f16 (LDS-staged, no f32 HBM rt)
//   EPI=2: multiply elementwise with `other` (may alias outF), write f32
// Block: 256 thr = 8 waves; 64 rows x 128 cols per block; wave w handles
// row-tile (w>>1), column half (w&1): 4 accum tiles, 16 v_wmma per wave.
// ---------------------------------------------------------------------------
template<int EPI>
__global__ __launch_bounds__(256) void gemm_mlp(
    const _Float16* __restrict__ A, const _Float16* __restrict__ Wt,
    const float* __restrict__ bias, const float* __restrict__ gamma,
    const float* __restrict__ beta, const float* other,
    float* outF, _Float16* __restrict__ outH, int nrows)
{
  __shared__ _Float16 Wsh[DIMF * WLD];
  const int tid  = threadIdx.x;
  const int lane = tid & 31;
  const int wave = tid >> 5;
  const int row0 = blockIdx.x * 64;

  __builtin_prefetch(A + (size_t)row0 * DIMF, 0, 0);   // global_prefetch_b8

  // Stage W^T (f16, 32KB) into padded LDS: 2048 x 16B, 8 per thread.
  #pragma unroll
  for (int j = 0; j < 8; ++j) {
    int e = tid + j * 256;
    int r = e >> 4, c = (e & 15) * 8;
    *(uint4*)(&Wsh[r * WLD + c]) = *(const uint4*)(&Wt[r * DIMF + c]);
  }
  __syncthreads();

  const int rt = wave >> 1;              // row tile 0..3
  const int cg = wave & 1;               // column half 0..1
  const int arow0 = row0 + rt * 16;

  v8f acc[4] = {};
  #pragma unroll
  for (int kk = 0; kk < 4; ++kk) {       // K = 128 in 4 steps of 32
    const int k0 = kk * 32;
    v16h a = load_a_g(A, arow0, k0, nrows, lane);
    #pragma unroll
    for (int t = 0; t < 4; ++t) {
      v16h b = load_b_lds(Wsh, cg * 64 + t * 16, k0, lane);
      acc[t] = __builtin_amdgcn_wmma_f32_16x16x32_f16(
          false, a, false, b, (short)0, acc[t], false, false);
    }
  }

  if (EPI == 1) {
    // ---- fused LayerNorm + Mish, f16 output ----
    __syncthreads();                       // all waves done reading weights
    float* st = (float*)Wsh;               // reuse LDS: 64 x 128 f32 staging
    #pragma unroll
    for (int t = 0; t < 4; ++t) {
      int col = cg * 64 + t * 16 + (lane & 15);
      #pragma unroll
      for (int r = 0; r < 8; ++r) {
        int m = rt * 16 + (lane >> 4) * 8 + r;   // D-tile VGPR layout
        st[m * DIMF + col] = acc[t][r] + bias[col];
      }
    }
    __syncthreads();
    const int r = tid >> 2, q = tid & 3;   // 4 threads per row, same wave
    const float* rowp = st + r * DIMF + q * 32;
    float s = 0.f, sq = 0.f;
    #pragma unroll
    for (int c = 0; c < 32; ++c) { float v = rowp[c]; s += v; sq += v * v; }
    s += __shfl_xor(s, 1);  sq += __shfl_xor(sq, 1);
    s += __shfl_xor(s, 2);  sq += __shfl_xor(sq, 2);
    const float mean = s * (1.f / 128.f);
    const float var  = sq * (1.f / 128.f) - mean * mean;
    const float rstd = rsqrtf(var + 1e-5f);
    const int grow = row0 + r;
    if (grow < nrows) {
      #pragma unroll
      for (int c = 0; c < 32; ++c) {
        int col = q * 32 + c;
        float v = (rowp[c] - mean) * rstd * gamma[col] + beta[col];
        float sp = (v > 20.f) ? v : log1pf(__expf(v));   // softplus
        outH[(size_t)grow * DIMF + col] = (_Float16)(v * tanhf(sp));  // mish
      }
    }
  } else {
    // ---- plain / product epilogue, f32 output ----
    #pragma unroll
    for (int t = 0; t < 4; ++t) {
      int col = cg * 64 + t * 16 + (lane & 15);
      float bcol = bias[col];
      #pragma unroll
      for (int r = 0; r < 8; ++r) {
        int row = row0 + rt * 16 + (lane >> 4) * 8 + r;
        if (row < nrows) {
          size_t idx = (size_t)row * DIMF + col;
          float v = acc[t][r] + bcol;
          if (EPI == 2) v *= other[idx];      // y = kOut * vOut (in place ok)
          outF[idx] = v;
        }
      }
    }
  }
}

// ---------------------------------------------------------------------------
// Helpers
// ---------------------------------------------------------------------------
__global__ void cvt_f16(const float* __restrict__ in, _Float16* __restrict__ out, int n) {
  int i = blockIdx.x * blockDim.x + threadIdx.x;
  int stride = gridDim.x * blockDim.x;
  for (; i < n; i += stride) out[i] = (_Float16)in[i];
}

// W[k][n] f32 -> Wt[n][k] f16 (column-major weights = B-fragment order)
__global__ void wt_cvt(const float* __restrict__ W, _Float16* __restrict__ Wt) {
  int n = blockIdx.x, k = threadIdx.x;
  Wt[n * DIMF + k] = (_Float16)W[k * DIMF + n];
}

// starts[b] = lower_bound(batch, b); batch is sorted. starts[NB] == N.
__global__ void seg_starts(const int* __restrict__ batch, int* __restrict__ starts,
                           int n, int nb) {
  int b = blockIdx.x * blockDim.x + threadIdx.x;
  if (b > nb) return;
  int lo = 0, hi = n;
  while (lo < hi) { int mid = (lo + hi) >> 1; if (batch[mid] < b) lo = mid + 1; else hi = mid; }
  starts[b] = lo;
}

// Per-segment, per-feature softmax over the node axis. One block per segment,
// thread f owns feature column f (coalesced across the 128 threads).
__global__ void seg_softmax_k(const float* __restrict__ key0,
                              const int* __restrict__ starts,
                              _Float16* __restrict__ keys16) {
  const int b = blockIdx.x, f = threadIdx.x;
  const int s = starts[b], e = starts[b + 1];
  float m = -3.4e38f;
  for (int i = s; i < e; ++i) m = fmaxf(m, key0[(size_t)i * DIMF + f]);
  float d = 0.f;
  for (int i = s; i < e; ++i) d += __expf(key0[(size_t)i * DIMF + f] - m);
  const float inv = (d > 0.f) ? 1.f / d : 0.f;
  for (int i = s; i < e; ++i)
    keys16[(size_t)i * DIMF + f] = (_Float16)(__expf(key0[(size_t)i * DIMF + f] - m) * inv);
}

// Deterministic segment sum (batch sorted, one block per segment).
__global__ void seg_sum(const float* __restrict__ y, const int* __restrict__ starts,
                        float* __restrict__ out) {
  const int b = blockIdx.x, f = threadIdx.x;
  const int s = starts[b], e = starts[b + 1];
  float acc = 0.f;
  for (int i = s; i < e; ++i) acc += y[(size_t)i * DIMF + f];
  out[(size_t)b * DIMF + f] = acc;
}

// ---------------------------------------------------------------------------
extern "C" void kernel_launch(void* const* d_in, const int* in_sizes, int n_in,
                              void* d_out, int out_size, void* d_ws, size_t ws_size,
                              hipStream_t stream) {
  (void)n_in; (void)out_size; (void)ws_size;
  const int N  = in_sizes[0] / DIMF;
  const int NB = NBATCH;

  const float* x     = (const float*)d_in[0];
  const int*   batch = (const int*)d_in[1];
  // weights / params (f32)
  const float* W_map = (const float*)d_in[3];  const float* b_map = (const float*)d_in[4];
  const float* kW1 = (const float*)d_in[5];  const float* kb1 = (const float*)d_in[6];
  const float* kg  = (const float*)d_in[7];  const float* kbn = (const float*)d_in[8];
  const float* kW2 = (const float*)d_in[9];  const float* kb2 = (const float*)d_in[10];
  const float* vW1 = (const float*)d_in[11]; const float* vb1 = (const float*)d_in[12];
  const float* vg  = (const float*)d_in[13]; const float* vbn = (const float*)d_in[14];
  const float* vW2 = (const float*)d_in[15]; const float* vb2 = (const float*)d_in[16];
  const float* iW_map = (const float*)d_in[17]; const float* ib_map = (const float*)d_in[18];
  const float* ikW1 = (const float*)d_in[19]; const float* ikb1 = (const float*)d_in[20];
  const float* ikg  = (const float*)d_in[21]; const float* ikbn = (const float*)d_in[22];
  const float* ikW2 = (const float*)d_in[23]; const float* ikb2 = (const float*)d_in[24];
  const float* ivW1 = (const float*)d_in[25]; const float* ivb1 = (const float*)d_in[26];
  const float* ivg  = (const float*)d_in[27]; const float* ivbn = (const float*)d_in[28];
  const float* ivW2 = (const float*)d_in[29]; const float* ivb2 = (const float*)d_in[30];

  // ---- workspace carve-up ----
  char* ws = (char*)d_ws;
  const size_t fsz = (size_t)N * DIMF * sizeof(float);
  const size_t hsz = (size_t)N * DIMF * sizeof(_Float16);
  size_t off = 0;
  float*    bufA   = (float*)(ws + off);    off += fsz;   // key0 / key0b / kh / scratch
  float*    bufB   = (float*)(ws + off);    off += fsz;   // vOut -> y (in place)
  _Float16* x16    = (_Float16*)(ws + off); off += hsz;   // x f16; later keys2 f16
  _Float16* keys16 = (_Float16*)(ws + off); off += hsz;
  _Float16* h16    = (_Float16*)(ws + off); off += hsz;   // hidden activations f16
  _Float16* wt     = (_Float16*)(ws + off); off += 10 * (size_t)DIMF * DIMF * sizeof(_Float16);
  int*      starts = (int*)(ws + off);      off += (size_t)(NB + 1) * sizeof(int);
  _Float16* keys2_16 = x16;                 // x16 is dead once z is formed

  const int SL = DIMF * DIMF;
  const int gridG = (N + 63) / 64;
  float* z      = (float*)d_out;
  float* keyenc = z + (size_t)NB * DIMF;

  // ---- prep: f16 inputs, f16 transposed weights, segment ranges ----
  cvt_f16<<<2048, 256, 0, stream>>>(x, x16, N * DIMF);
  wt_cvt<<<128, 128, 0, stream>>>(W_map,  wt + 0 * SL);
  wt_cvt<<<128, 128, 0, stream>>>(kW1,    wt + 1 * SL);
  wt_cvt<<<128, 128, 0, stream>>>(kW2,    wt + 2 * SL);
  wt_cvt<<<128, 128, 0, stream>>>(vW1,    wt + 3 * SL);
  wt_cvt<<<128, 128, 0, stream>>>(vW2,    wt + 4 * SL);
  wt_cvt<<<128, 128, 0, stream>>>(iW_map, wt + 5 * SL);
  wt_cvt<<<128, 128, 0, stream>>>(ikW1,   wt + 6 * SL);
  wt_cvt<<<128, 128, 0, stream>>>(ikW2,   wt + 7 * SL);
  wt_cvt<<<128, 128, 0, stream>>>(ivW1,   wt + 8 * SL);
  wt_cvt<<<128, 128, 0, stream>>>(ivW2,   wt + 9 * SL);
  seg_starts<<<(NB + 1 + 255) / 256, 256, 0, stream>>>(batch, starts, N, NB);

  // ---- stage 1: keys = seg_softmax(x @ W_map + b_map) ----
  gemm_mlp<0><<<gridG, 256, 0, stream>>>(x16, wt + 0 * SL, b_map,
      nullptr, nullptr, nullptr, bufA, nullptr, N);
  seg_softmax_k<<<NB, 128, 0, stream>>>(bufA, starts, keys16);

  // ---- stage 2: z = seg_sum( val_net(x) * key_net(keys) ) ----
  gemm_mlp<1><<<gridG, 256, 0, stream>>>(x16, wt + 3 * SL, vb1,
      vg, vbn, nullptr, nullptr, h16, N);                       // LN+mish
  gemm_mlp<0><<<gridG, 256, 0, stream>>>(h16, wt + 4 * SL, vb2,
      nullptr, nullptr, nullptr, bufB, nullptr, N);             // vOut
  gemm_mlp<1><<<gridG, 256, 0, stream>>>(keys16, wt + 1 * SL, kb1,
      kg, kbn, nullptr, nullptr, h16, N);
  gemm_mlp<2><<<gridG, 256, 0, stream>>>(h16, wt + 2 * SL, kb2,
      nullptr, nullptr, bufB, bufB, nullptr, N);                // y = kOut*vOut
  seg_sum<<<NB, 128, 0, stream>>>(bufB, starts, z);

  // ---- stage 3: keys2 = seg_softmax(keys @ iW_map + ib_map) ----
  gemm_mlp<0><<<gridG, 256, 0, stream>>>(keys16, wt + 5 * SL, ib_map,
      nullptr, nullptr, nullptr, bufA, nullptr, N);
  seg_softmax_k<<<NB, 128, 0, stream>>>(bufA, starts, keys2_16);

  // ---- stage 4: keyenc = seg_sum( iv_mlp(keys) * ik_mlp(keys2) ) ----
  gemm_mlp<1><<<gridG, 256, 0, stream>>>(keys16, wt + 8 * SL, ivb1,
      ivg, ivbn, nullptr, nullptr, h16, N);
  gemm_mlp<0><<<gridG, 256, 0, stream>>>(h16, wt + 9 * SL, ivb2,
      nullptr, nullptr, nullptr, bufB, nullptr, N);             // ivOut
  gemm_mlp<1><<<gridG, 256, 0, stream>>>(keys2_16, wt + 6 * SL, ikb1,
      ikg, ikbn, nullptr, nullptr, h16, N);
  gemm_mlp<2><<<gridG, 256, 0, stream>>>(h16, wt + 7 * SL, ikb2,
      nullptr, nullptr, bufB, bufB, nullptr, N);                // y2
  seg_sum<<<NB, 128, 0, stream>>>(bufB, starts, keyenc);
}
